// StructureEnhancementBlock_16372415332881
// MI455X (gfx1250) — matrix operand order repaired and checked by reading, not compiled, not run on hardware
//
#include <hip/hip_runtime.h>

typedef __bf16 bf16_t;
typedef __attribute__((ext_vector_type(16))) __bf16 bf16x16;
typedef __attribute__((ext_vector_type(8)))  __bf16 bf16x8;
typedef __attribute__((ext_vector_type(8)))  float  v8f;

constexpr int kHW = 4096;   // 64*64
constexpr int kC  = 256;
constexpr int kB  = 16;
constexpr int kH  = 64;
constexpr int kW  = 64;

// ---------------- WMMA helpers (gfx1250 wave32) ----------------

__device__ __forceinline__ v8f wmma_bf16(bf16x16 a, bf16x16 b, v8f c) {
  return __builtin_amdgcn_wmma_f32_16x16x32_bf16(false, a, false, b, (short)0, c,
                                                 false, false);
}

// A 16x32 bf16 fragment: lane row = lane%16, khalf = lane/16.
// elements 0..7  <-> K = khalf*8 + 0..7
// elements 8..15 <-> K = 16 + khalf*8 + 0..7
__device__ __forceinline__ bf16x16 load_a_frag(const bf16_t* row, int khalf) {
  bf16x8 c0 = *(const bf16x8*)(row + khalf * 8);
  bf16x8 c1 = *(const bf16x8*)(row + 16 + khalf * 8);
  bf16x16 a;
#pragma unroll
  for (int e = 0; e < 8; ++e) { a[e] = c0[e]; a[e + 8] = c1[e]; }
  return a;
}

// B 32x16 bf16 fragment: lane col = lane%16, khalf = lane/16.
// elements 0..7  <-> K = khalf*8 + 0..7
// elements 8..15 <-> K = 16 + khalf*8 + 0..7
__device__ __forceinline__ bf16x16 load_b_frag(const bf16_t* row, int khalf) {
  bf16x8 c0 = *(const bf16x8*)(row + khalf * 8);
  bf16x8 c1 = *(const bf16x8*)(row + 16 + khalf * 8);
  bf16x16 b;
#pragma unroll
  for (int e = 0; e < 8; ++e) { b[e] = c0[e]; b[e + 8] = c1[e]; }
  return b;
}

// ---------------- weight fp32 -> bf16 ----------------

__global__ __launch_bounds__(256) void cvt_kernel(const float* __restrict__ src,
                                                  bf16_t* __restrict__ dst, int n) {
  int i = blockIdx.x * 256 + threadIdx.x;
  if (i < n) dst[i] = (bf16_t)src[i];
}

// ---------------- LayerNorm over channels, output transposed bf16 ----------------
// in:  X [N][C=256][HW] fp32 ; out: OutT [N][HW][C] bf16 (row = pixel, contiguous C)
__global__ __launch_bounds__(256) void ln_kernel(const float* __restrict__ X,
                                                 const float* __restrict__ gamma,
                                                 const float* __restrict__ beta,
                                                 bf16_t* __restrict__ OutT) {
  __shared__ float tile[256][33];
  const int n   = blockIdx.y;
  const int hw0 = blockIdx.x * 32;
  const int tid = threadIdx.x;
  const int cpart = tid >> 5;   // 0..7
  const int col   = tid & 31;   // 0..31
#pragma unroll 4
  for (int it = 0; it < 32; ++it) {
    int c = it * 8 + cpart;
    tile[c][col] = X[((size_t)n * kC + c) * kHW + hw0 + col];
  }
  __syncthreads();
  const int wave = tid >> 5, lane = tid & 31;
#pragma unroll
  for (int pi = 0; pi < 4; ++pi) {
    int p = wave * 4 + pi;
    float s = 0.f, s2 = 0.f;
#pragma unroll
    for (int i = 0; i < 8; ++i) {
      float v = tile[lane + i * 32][p];
      s += v; s2 += v * v;
    }
#pragma unroll
    for (int off = 16; off > 0; off >>= 1) {
      s  += __shfl_xor(s,  off, 32);
      s2 += __shfl_xor(s2, off, 32);
    }
    float u   = s * (1.f / 256.f);
    float var = s2 * (1.f / 256.f) - u * u;
    float r   = rsqrtf(var + 1e-6f);
    bf16x8 ov;
#pragma unroll
    for (int i = 0; i < 8; ++i) {
      int c = lane * 8 + i;
      ov[i] = (bf16_t)((tile[c][p] - u) * r * gamma[c] + beta[c]);
    }
    *(bf16x8*)(OutT + ((size_t)n * kHW + hw0 + p) * kC + lane * 8) = ov;
  }
}

// ---------------- generic C=256-K GEMM: D[b][m][n] = sum_k W[m][k] * XT[b][n][k] ----------------
enum { GEMM_Q = 0, GEMM_KV = 1, GEMM_O = 2, GEMM_FC1 = 3, GEMM_FC2 = 4 };

template <int MODE>
__global__ __launch_bounds__(256) void gemm_kernel(
    const bf16_t* __restrict__ Wb,    // [M][256] bf16
    const bf16_t* __restrict__ XT,    // [batch][HW][256] bf16
    float*  __restrict__ OutF,        // fp32 row-major out (O/FC1/FC2)
    bf16_t* __restrict__ OutB,        // bf16 row-major out (Q, KV k-part)
    bf16_t* __restrict__ OutT,        // bf16 transposed out (KV v-part)
    const float* __restrict__ Res,    // residual (O: img_feat, FC2: x)
    const float* __restrict__ Bias) { // o_proj bias
  const int K = 256;
  const int batch = blockIdx.z;
  const int n_blk = blockIdx.x * 128;
  const int m_blk = blockIdx.y * 64;
  const int tid = threadIdx.x;
  const int wave = tid >> 5, lane = tid & 31;
  const int wm = wave & 3, wn = wave >> 2;
  const int m0 = m_blk + wm * 16;
  const int nbase = n_blk + wn * 64;
  const int khalf = lane >> 4, lmod = lane & 15;

  const bf16_t* arow  = Wb + (size_t)(m0 + lmod) * K;
  const bf16_t* xbase = XT + (size_t)batch * kHW * K;

  v8f acc[4] = {v8f{}, v8f{}, v8f{}, v8f{}};
#pragma unroll 2
  for (int k0 = 0; k0 < K; k0 += 32) {
    bf16x16 a = load_a_frag(arow + k0, khalf);
#pragma unroll
    for (int t = 0; t < 4; ++t) {
      const bf16_t* xrow = xbase + (size_t)(nbase + t * 16 + lmod) * K + k0;
      bf16x16 bf = load_b_frag(xrow, khalf);
      acc[t] = wmma_bf16(a, bf, acc[t]);
    }
  }

  const int rowbase = m0 + khalf * 8;  // D row = rowbase + v, col = n + lmod
  if (MODE == GEMM_Q) {
#pragma unroll
    for (int t = 0; t < 4; ++t) {
      int coln = nbase + t * 16 + lmod;
#pragma unroll
      for (int v = 0; v < 8; ++v)
        OutB[((size_t)batch * kC + rowbase + v) * kHW + coln] = (bf16_t)acc[t][v];
    }
  } else if (MODE == GEMM_KV) {
    if (rowbase < 256) {  // k-part, row-major [2B][256][HW]
#pragma unroll
      for (int t = 0; t < 4; ++t) {
        int coln = nbase + t * 16 + lmod;
#pragma unroll
        for (int v = 0; v < 8; ++v)
          OutB[((size_t)batch * kC + rowbase + v) * kHW + coln] = (bf16_t)acc[t][v];
      }
    } else {              // v-part, transposed [2B][HW][256]
#pragma unroll
      for (int t = 0; t < 4; ++t) {
        int coln = nbase + t * 16 + lmod;
        bf16x8 ov;
#pragma unroll
        for (int v = 0; v < 8; ++v) ov[v] = (bf16_t)acc[t][v];
        *(bf16x8*)(OutT + ((size_t)batch * kHW + coln) * kC + (rowbase - 256)) = ov;
      }
    }
  } else if (MODE == GEMM_O) {
#pragma unroll
    for (int t = 0; t < 4; ++t) {
      int coln = nbase + t * 16 + lmod;
#pragma unroll
      for (int v = 0; v < 8; ++v) {
        size_t off = ((size_t)batch * kC + rowbase + v) * kHW + coln;
        OutF[off] = acc[t][v] + Res[off] + Bias[rowbase + v];
      }
    }
  } else if (MODE == GEMM_FC1) {
#pragma unroll
    for (int t = 0; t < 4; ++t) {
      int coln = nbase + t * 16 + lmod;
#pragma unroll
      for (int v = 0; v < 8; ++v)
        OutF[((size_t)batch * kC + rowbase + v) * kHW + coln] = acc[t][v];
    }
  } else {  // GEMM_FC2: out = acc + x
#pragma unroll
    for (int t = 0; t < 4; ++t) {
      int coln = nbase + t * 16 + lmod;
#pragma unroll
      for (int v = 0; v < 8; ++v) {
        size_t off = ((size_t)batch * kC + rowbase + v) * kHW + coln;
        OutF[off] = acc[t][v] + Res[off];
      }
    }
  }
}

// ---------------- row L2-norm inverse: inv[row] = 1/max(||x||,1e-12), rows of 4096 ----------------
__global__ __launch_bounds__(256) void rownorm_kernel(const bf16_t* __restrict__ X,
                                                      float* __restrict__ inv) {
  __shared__ float red[256];
  const int row = blockIdx.x;
  const int tid = threadIdx.x;
  const bf16_t* p = X + (size_t)row * kHW;
  float s = 0.f;
  for (int i = tid; i < kHW; i += 256) { float v = (float)p[i]; s += v * v; }
  red[tid] = s;
  __syncthreads();
  for (int off = 128; off > 0; off >>= 1) {
    if (tid < off) red[tid] += red[tid + off];
    __syncthreads();
  }
  if (tid == 0) inv[row] = 1.f / fmaxf(sqrtf(red[0]), 1e-12f);
}

// ---------------- Gram (over HW) + softmax, P -> global ----------------
// Q: [B][256][HW] bf16, Km: [2B][256][HW] bf16, Pout: [B*NH][2][64][64] fp32
__global__ __launch_bounds__(256) void gram_softmax_kernel(
    const bf16_t* __restrict__ Q, const bf16_t* __restrict__ Km,
    const float* __restrict__ qinv, const float* __restrict__ kinv,
    float* __restrict__ Pout) {
  __shared__ float P[2][64][65];
  const int bh = blockIdx.x, b = bh >> 2, nh = bh & 3, cbase = nh * 64;
  const int tid = threadIdx.x, wave = tid >> 5, lane = tid & 31;
  const int khalf = lane >> 4, lmod = lane & 15;

  for (int job = wave; job < 32; job += 8) {
    int br = job >> 4, tile = job & 15, tm = tile >> 2, tn = tile & 3;
    int ab = br * kB + b;
    const bf16_t* qrow = Q  + ((size_t)b  * kC + cbase + tm * 16 + lmod) * kHW;
    const bf16_t* krow = Km + ((size_t)ab * kC + cbase + tn * 16 + lmod) * kHW;
    v8f acc = {};
#pragma unroll 4
    for (int k0 = 0; k0 < kHW; k0 += 32)
      acc = wmma_bf16(load_a_frag(qrow + k0, khalf), load_b_frag(krow + k0, khalf), acc);
    int rc = tm * 16 + khalf * 8, cd = tn * 16 + lmod;
    float kn = kinv[ab * kC + cbase + cd];
#pragma unroll
    for (int v = 0; v < 8; ++v)
      P[br][rc + v][cd] = acc[v] * qinv[b * kC + cbase + rc + v] * kn;
  }
  __syncthreads();
  if (tid < 128) {  // softmax over d per (branch,row)
    int br = tid >> 6, r = tid & 63;
    float mx = -1e30f;
    for (int d = 0; d < 64; ++d) mx = fmaxf(mx, P[br][r][d]);
    float s = 0.f;
    for (int d = 0; d < 64; ++d) { float e = __expf(P[br][r][d] - mx); P[br][r][d] = e; s += e; }
    float is = 1.f / s;
    for (int d = 0; d < 64; ++d) P[br][r][d] *= is;
  }
  __syncthreads();
  float* dst = Pout + (size_t)bh * 2 * 64 * 64;
  for (int i = tid; i < 2 * 64 * 64; i += 256) {
    int br = i >> 12, rem = i & 4095;
    dst[i] = P[br][rem >> 6][rem & 63];
  }
}

// ---------------- attention apply: qkvT[b][n][c] = (g*P_tex*V_tex + (1-g)*P_dep*V_dep)^T ----------------
__global__ __launch_bounds__(256) void attn_apply_kernel(
    const bf16_t* __restrict__ VT,      // [2B][HW][256] bf16
    const float* __restrict__ Pin,      // [B*NH][2][64][64]
    const float* __restrict__ attn_scale,
    bf16_t* __restrict__ qkvT) {        // [B][HW][256] bf16
  __shared__ float P[2][64][65];
  const int bh = blockIdx.x, b = bh >> 2, nh = bh & 3, cbase = nh * 64;
  const int tid = threadIdx.x, wave = tid >> 5, lane = tid & 31;
  const int khalf = lane >> 4, lmod = lane & 15;
  const float* src = Pin + (size_t)bh * 2 * 64 * 64;
  for (int i = tid; i < 2 * 64 * 64; i += 256) {
    int br = i >> 12, rem = i & 4095;
    P[br][rem >> 6][rem & 63] = src[i];
  }
  __syncthreads();
  const float g = 1.f / (1.f + __expf(-attn_scale[nh]));
  const int nbase = blockIdx.y * 512;
  // jobs: 32 n-tiles x 4 c-tiles
  for (int job = wave; job < 128; job += 8) {
    int nt = job >> 2, ct = job & 3;
    int n0 = nbase + nt * 16;
    v8f accs[2];
#pragma unroll
    for (int br = 0; br < 2; ++br) {
      int ab = br * kB + b;
      v8f acc = {};
#pragma unroll
      for (int ks = 0; ks < 2; ++ks) {
        int k0 = ks * 32;
        bf16x16 a;
        const float* prow = &P[br][ct * 16 + lmod][0];
#pragma unroll
        for (int e = 0; e < 8; ++e) {
          a[e]     = (bf16_t)prow[k0 + khalf * 8 + e];
          a[e + 8] = (bf16_t)prow[k0 + 16 + khalf * 8 + e];
        }
        const bf16_t* vrow = VT + ((size_t)ab * kHW + n0 + lmod) * kC + cbase + k0;
        bf16x16 bf = load_b_frag(vrow, khalf);
        acc = wmma_bf16(a, bf, acc);
      }
      accs[br] = acc;
    }
    bf16x8 ov;
#pragma unroll
    for (int v = 0; v < 8; ++v)
      ov[v] = (bf16_t)(g * accs[0][v] + (1.f - g) * accs[1][v]);
    *(bf16x8*)(qkvT + ((size_t)b * kHW + n0 + lmod) * kC + cbase + ct * 16 + khalf * 8) = ov;
  }
}

// ---------------- depthwise 3x3 (SAME, zero pad) + exact GELU, transposed bf16 out ----------------
__global__ __launch_bounds__(256) void dwgelu_kernel(const float* __restrict__ Y,
                                                     const float* __restrict__ wdw,
                                                     bf16_t* __restrict__ OutT) {
  int idx = blockIdx.x * 256 + threadIdx.x;        // over B*C*HW = 16.7M
  int hw = idx & (kHW - 1);
  int c  = (idx >> 12) & (kC - 1);
  int b  = idx >> 20;
  int h = hw >> 6, x = hw & 63;
  const float* base = Y + ((size_t)b * kC + c) * kHW;
  const float* wc = wdw + c * 9;
  float acc = 0.f;
#pragma unroll
  for (int dy = -1; dy <= 1; ++dy) {
#pragma unroll
    for (int dx = -1; dx <= 1; ++dx) {
      int hh = h + dy, xx = x + dx;
      if (hh >= 0 && hh < kH && xx >= 0 && xx < kW)
        acc += wc[(dy + 1) * 3 + (dx + 1)] * base[hh * kW + xx];
    }
  }
  float ge = 0.5f * acc * (1.f + erff(acc * 0.70710678118654752f));
  OutT[((size_t)b * kHW + hw) * kC + c] = (bf16_t)ge;
}

// ---------------- launch ----------------
extern "C" void kernel_launch(void* const* d_in, const int* in_sizes, int n_in,
                              void* d_out, int out_size, void* d_ws, size_t ws_size,
                              hipStream_t stream) {
  const float* img      = (const float*)d_in[0];
  const float* aux_tex  = (const float*)d_in[1];
  const float* aux_dep  = (const float*)d_in[2];
  const float* q_nw     = (const float*)d_in[3];
  const float* q_nb     = (const float*)d_in[4];
  const float* kv_nw    = (const float*)d_in[5];
  const float* kv_nb    = (const float*)d_in[6];
  const float* attn_sc  = (const float*)d_in[7];
  const float* q_w      = (const float*)d_in[8];
  const float* kv_w     = (const float*)d_in[9];
  const float* o_w      = (const float*)d_in[10];
  const float* o_b      = (const float*)d_in[11];
  const float* ffn_nw   = (const float*)d_in[12];
  const float* ffn_nb   = (const float*)d_in[13];
  const float* fc1_w    = (const float*)d_in[14];
  const float* dw_w     = (const float*)d_in[15];
  const float* fc2_w    = (const float*)d_in[16];
  float* out = (float*)d_out;

  char* ws = (char*)d_ws;
  size_t off = 0;
  auto take = [&](size_t bytes) { size_t o = off; off += (bytes + 255) & ~(size_t)255; return o; };
  const size_t act_q  = (size_t)kB * kHW * kC;       // 16.7M elems
  const size_t act_kv = 2 * act_q;

  size_t wq_off  = take(65536 * 2);
  size_t wkv_off = take(131072 * 2);
  size_t wo_off  = take(65536 * 2);
  size_t wf1_off = take(65536 * 2);
  size_t wf2_off = take(65536 * 2);
  size_t xqt_off = take(act_q * 2);    // reused later as qkvT
  size_t xkvt_off= take(act_kv * 2);   // reused later as x (fp32, same 64MB)
  size_t qb_off  = take(act_q * 2);    // reused later as YT
  size_t kb_off  = take(act_kv * 2);   // reused later as Y1 (fp32)
  size_t vt_off  = take(act_kv * 2);   // reused later as Y2T
  size_t qi_off  = take((size_t)kB * kC * 4);
  size_t ki_off  = take((size_t)2 * kB * kC * 4);
  size_t p_off   = take((size_t)kB * 4 * 2 * 64 * 64 * 4);
  (void)ws_size; (void)n_in; (void)in_sizes; (void)out_size;

  bf16_t* WQ   = (bf16_t*)(ws + wq_off);
  bf16_t* WKV  = (bf16_t*)(ws + wkv_off);
  bf16_t* WO   = (bf16_t*)(ws + wo_off);
  bf16_t* WF1  = (bf16_t*)(ws + wf1_off);
  bf16_t* WF2  = (bf16_t*)(ws + wf2_off);
  bf16_t* XqT  = (bf16_t*)(ws + xqt_off);
  bf16_t* XkvT = (bf16_t*)(ws + xkvt_off);
  bf16_t* Qb   = (bf16_t*)(ws + qb_off);
  bf16_t* Kb   = (bf16_t*)(ws + kb_off);
  bf16_t* VTb  = (bf16_t*)(ws + vt_off);
  float*  qinv = (float*)(ws + qi_off);
  float*  kinv = (float*)(ws + ki_off);
  float*  Pbuf = (float*)(ws + p_off);
  bf16_t* qkvT = XqT;                       // alias (XqT dead after q-GEMM)
  float*  Xres = (float*)(ws + xkvt_off);   // alias (XkvT dead after kv-GEMM)
  bf16_t* YT   = Qb;                        // alias (Q dead after gram)
  float*  Y1   = (float*)(ws + kb_off);     // alias (K dead after gram)
  bf16_t* Y2T  = VTb;                       // alias (V dead after attn_apply)

  // 1) weights -> bf16
  cvt_kernel<<<256,  256, 0, stream>>>(q_w,   WQ,  65536);
  cvt_kernel<<<512,  256, 0, stream>>>(kv_w,  WKV, 131072);
  cvt_kernel<<<256,  256, 0, stream>>>(o_w,   WO,  65536);
  cvt_kernel<<<256,  256, 0, stream>>>(fc1_w, WF1, 65536);
  cvt_kernel<<<256,  256, 0, stream>>>(fc2_w, WF2, 65536);

  // 2) LayerNorms -> transposed bf16 activations
  dim3 lng(kHW / 32, kB);
  ln_kernel<<<lng, 256, 0, stream>>>(img,     q_nw,  q_nb,  XqT);
  ln_kernel<<<lng, 256, 0, stream>>>(aux_tex, kv_nw, kv_nb, XkvT);
  ln_kernel<<<lng, 256, 0, stream>>>(aux_dep, kv_nw, kv_nb, XkvT + act_q);

  // 3) projections
  gemm_kernel<GEMM_Q ><<<dim3(32, 4, 16), 256, 0, stream>>>(WQ,  XqT,  nullptr, Qb, nullptr, nullptr, nullptr);
  gemm_kernel<GEMM_KV><<<dim3(32, 8, 32), 256, 0, stream>>>(WKV, XkvT, nullptr, Kb, VTb,     nullptr, nullptr);

  // 4) spatial L2 norms of q and k
  rownorm_kernel<<<kB * kC,     256, 0, stream>>>(Qb, qinv);
  rownorm_kernel<<<2 * kB * kC, 256, 0, stream>>>(Kb, kinv);

  // 5) attention
  gram_softmax_kernel<<<kB * 4, 256, 0, stream>>>(Qb, Kb, qinv, kinv, Pbuf);
  attn_apply_kernel<<<dim3(kB * 4, 8), 256, 0, stream>>>(VTb, Pbuf, attn_sc, qkvT);

  // 6) o-proj + residual -> x (fp32)
  gemm_kernel<GEMM_O><<<dim3(32, 4, 16), 256, 0, stream>>>(WO, qkvT, Xres, nullptr, nullptr, img, o_b);

  // 7) FFN
  ln_kernel<<<lng, 256, 0, stream>>>(Xres, ffn_nw, ffn_nb, YT);
  gemm_kernel<GEMM_FC1><<<dim3(32, 4, 16), 256, 0, stream>>>(WF1, YT, Y1, nullptr, nullptr, nullptr, nullptr);
  dwgelu_kernel<<<(kB * kC * kHW) / 256, 256, 0, stream>>>(Y1, dw_w, Y2T);
  gemm_kernel<GEMM_FC2><<<dim3(32, 4, 16), 256, 0, stream>>>(WF2, Y2T, out, nullptr, nullptr, Xres, nullptr);
}
